// Bah_attention_163208757668
// MI455X (gfx1250) — compile-verified, matrix operand order checked by reading, not run on hardware
//
#include <hip/hip_runtime.h>
#include <cmath>

// Problem constants (match reference)
#define BATCH 32
#define TLEN  2048
#define HDIM  1024

typedef __attribute__((ext_vector_type(16))) __bf16         v16bf;
typedef __attribute__((ext_vector_type(8)))  float          v8f;
typedef __attribute__((ext_vector_type(16))) unsigned short v16u;
typedef __attribute__((ext_vector_type(8)))  unsigned short v8u;
typedef __attribute__((ext_vector_type(4)))  unsigned int   u32x4;
typedef __attribute__((ext_vector_type(8)))  int            i32x8;
typedef __attribute__((ext_vector_type(4)))  int            i32x4;

// ---- Tensor Data Mover availability / arity detection -----------------------
#if defined(__gfx1250__) && __has_builtin(__builtin_amdgcn_tensor_load_to_lds)
#define HAVE_TDM 1
#if __has_include(<hip/amd_detail/amd_gfx1250_TDM.h>)
#define TDM_SIX_ARGS 1      // amdgpu-toolchain (therock-10.0 headers): 6-arg form
#else
#define TDM_SIX_ARGS 0      // ROCm 7.2 / clang-22: 5-arg form
#endif
#else
#define HAVE_TDM 0
#endif

__device__ __forceinline__ unsigned short f2bf(float f) {
    unsigned int u = __builtin_bit_cast(unsigned int, f);
    u += 0x7FFFu + ((u >> 16) & 1u);          // round-to-nearest-even
    return (unsigned short)(u >> 16);
}

#if HAVE_TDM
// Issue a TDM 2D-tile load (global -> LDS).  D# packed per CDNA5 ISA §8:
//  group0: count=1 | lds_addr | global_addr[56:0] | type=2
//  group1: mask=0, data_size=4B, tensor dims/strides, tile dims (elements)
__device__ __forceinline__ void tdm_load_2d(unsigned lds_addr, const void* gptr,
                                            unsigned tile_d0, unsigned tile_d1,
                                            unsigned tensor_d0, unsigned tensor_d1,
                                            unsigned stride0) {
    unsigned long long ga = (unsigned long long)gptr;
    u32x4 g0;
    g0[0] = 1u;                                             // count=1 (user D#)
    g0[1] = lds_addr;                                       // bits 63:32
    g0[2] = (unsigned)(ga & 0xFFFFFFFFu);                   // bits 95:64
    g0[3] = (unsigned)((ga >> 32) & 0x1FFFFFFu) | (2u << 30); // addr hi | type=2
    i32x8 g1;
    g1[0] = (int)(2u << 16);                                // data_size=4B, mask=0
    g1[1] = (int)((tensor_d0 & 0xFFFFu) << 16);             // tensor_dim0 lo
    g1[2] = (int)(((tensor_d0 >> 16) & 0xFFFFu) |
                  ((tensor_d1 & 0xFFFFu) << 16));           // dim0 hi | dim1 lo
    g1[3] = (int)(((tensor_d1 >> 16) & 0xFFFFu) |
                  (tile_d0 << 16));                         // dim1 hi | tile_dim0
    g1[4] = (int)(tile_d1 & 0xFFFFu);                       // tile_dim1 | tile_dim2=0
    g1[5] = (int)stride0;                                   // tensor_dim0_stride lo
    g1[6] = 0;                                              // stride hi | dim1_stride
    g1[7] = 0;
    i32x4 z4 = {0, 0, 0, 0};
#if TDM_SIX_ARGS
    i32x8 z8 = {0, 0, 0, 0, 0, 0, 0, 0};
    __builtin_amdgcn_tensor_load_to_lds(g0, g1, z4, z4, z8, 0);
#else
    __builtin_amdgcn_tensor_load_to_lds(g0, g1, z4, z4, 0);
#endif
}
#endif

// ---------------------------------------------------------------------------
// Kernel 0: zero the raw-score accumulator.
// ---------------------------------------------------------------------------
__global__ __launch_bounds__(256)
void zero_kernel(float* __restrict__ p, int n) {
    int i = blockIdx.x * 256 + threadIdx.x;
    if (i < n) p[i] = 0.0f;
}

// ---------------------------------------------------------------------------
// Kernel 1: g[b,o] = sum_h x[b,h]*Wq[o,h] + bq[o] + bm[o]
// ---------------------------------------------------------------------------
__global__ __launch_bounds__(256)
void query_proj_kernel(const float* __restrict__ x, const float* __restrict__ Wq,
                       const float* __restrict__ bq, const float* __restrict__ bm,
                       float* __restrict__ g) {
    const int b    = blockIdx.y;
    const int lane = threadIdx.x & 31;
    const int wv   = threadIdx.x >> 5;
    const int o    = blockIdx.x * 8 + wv;
    const float* xr = x + (size_t)b * HDIM;
    const float* wr = Wq + (size_t)o * HDIM;
    float acc = 0.0f;
    for (int h = lane; h < HDIM; h += 32) acc += xr[h] * wr[h];
#pragma unroll
    for (int m = 16; m > 0; m >>= 1) acc += __shfl_xor(acc, m, 32);
    if (lane == 0) g[(size_t)b * HDIM + o] = acc + bq[o] + bm[o];
}

// ---------------------------------------------------------------------------
// Kernel 2: pack Wm (fp32 [o][k]) into bf16 WMMA B fragments (lane layout per
// ISA 16-bit operand tables) so each lane's B operand is one contiguous,
// coalesced 32B load.
// ---------------------------------------------------------------------------
__global__ __launch_bounds__(256)
void pack_wm_kernel(const float* __restrict__ Wm, unsigned short* __restrict__ wmP) {
    int idx  = blockIdx.x * 256 + threadIdx.x;  // ((nt*32+kc)*32+lane)*16+e
    int e    = idx & 15;
    int lane = (idx >> 4) & 31;
    int kc   = (idx >> 9) & 31;
    int nt   = idx >> 14;
    int o = nt * 16 + (lane & 15);
    int k = kc * 32 + ((e < 8) ? e : (e + 8)) + ((lane < 16) ? 0 : 8);
    wmP[idx] = f2bf(Wm[(size_t)o * HDIM + k]);
}

// ---------------------------------------------------------------------------
// Kernel 3: fused  hc = context @ Wm^T  (bf16 WMMA, f32 acc)  +
//           sraw[b,t] += sum_o tanh(hc + g) * v[o].
// Block = 64 rows x 256 cols, K=1024 in steps of 32.  A tile double-buffered
// in LDS, filled by the Tensor Data Mover (wave 0 issues; TENSORcnt gates),
// overlapping DMA with the 8-WMMA compute phase.
// ---------------------------------------------------------------------------
__global__ __launch_bounds__(256)
void gemm_tanh_kernel(const float* __restrict__ context,
                      const unsigned short* __restrict__ wmP,
                      const float* __restrict__ g,
                      const float* __restrict__ v,
                      float* __restrict__ sraw) {
    const int t0 = blockIdx.x * 64;
    const int o0 = blockIdx.y * 256;
    const int b  = blockIdx.z;

    const int lane = threadIdx.x & 31;
    const int wv   = threadIdx.x >> 5;
    const int mt   = wv & 3;        // M-tile within block (0..3)
    const int ngrp = wv >> 2;       // N half (0..1), 8 n-tiles each

    const int arow = mt * 16 + (lane & 15);
    const int koff = (lane < 16) ? 0 : 8;
    const int ntbase = (o0 >> 4) + ngrp * 8;
    constexpr int KCH = HDIM / 32;

    v8f acc[8] = {};

#if HAVE_TDM
    __shared__ float ldsAf[2][64 * 32];         // two 8KB fp32 tiles
    const float* gbase = context + ((size_t)b * TLEN + t0) * HDIM;
    const unsigned lds_off[2] = {(unsigned)(size_t)&ldsAf[0][0],
                                 (unsigned)(size_t)&ldsAf[1][0]};

    if (wv == 0)                                 // prime the pipeline
        tdm_load_2d(lds_off[0], gbase, /*tile*/32, 64,
                    /*tensor*/HDIM, 65536u, /*stride0*/HDIM);

    for (int kc = 0; kc < KCH; ++kc) {
        const int cur = kc & 1;
        if (wv == 0) {
            if (kc + 1 < KCH) {
                tdm_load_2d(lds_off[cur ^ 1], gbase + (kc + 1) * 32,
                            32, 64, HDIM, 65536u, HDIM);
                __builtin_amdgcn_s_wait_tensorcnt(1);  // current tile landed
            } else {
                __builtin_amdgcn_s_wait_tensorcnt(0);
            }
        }
        __syncthreads();

        // A fragment: 16 fp32 from LDS (K = {0..7,16..23}+koff), convert to bf16
        const float* ab = &ldsAf[cur][arow * 32 + koff];
        float4 f0 = *(const float4*)(ab + 0);
        float4 f1 = *(const float4*)(ab + 4);
        float4 f2 = *(const float4*)(ab + 16);
        float4 f3 = *(const float4*)(ab + 20);
        v16u au;
        au[0]  = f2bf(f0.x); au[1]  = f2bf(f0.y); au[2]  = f2bf(f0.z); au[3]  = f2bf(f0.w);
        au[4]  = f2bf(f1.x); au[5]  = f2bf(f1.y); au[6]  = f2bf(f1.z); au[7]  = f2bf(f1.w);
        au[8]  = f2bf(f2.x); au[9]  = f2bf(f2.y); au[10] = f2bf(f2.z); au[11] = f2bf(f2.w);
        au[12] = f2bf(f3.x); au[13] = f2bf(f3.y); au[14] = f2bf(f3.z); au[15] = f2bf(f3.w);
        v16bf af = __builtin_bit_cast(v16bf, au);

#pragma unroll
        for (int j = 0; j < 8; ++j) {
            const int ntg = ntbase + j;
            v16u bu = *(const v16u*)(wmP + ((size_t)(ntg * 32 + kc) * 32 + lane) * 16);
            v16bf bf = __builtin_bit_cast(v16bf, bu);
            acc[j] = __builtin_amdgcn_wmma_f32_16x16x32_bf16(
                false, af, false, bf, (short)0, acc[j], false, false);
        }
        __syncthreads();
    }
#else
    // Fallback: cooperative fp32->bf16 staging (proven path).
    __shared__ unsigned short ldsA[64 * 32];
    const int ldrow  = threadIdx.x >> 2;
    const int ldpart = threadIdx.x & 3;
    const float* src = context + ((size_t)b * TLEN + t0 + ldrow) * HDIM + ldpart * 8;

    for (int kc = 0; kc < KCH; ++kc) {
        float4 f0 = *(const float4*)(src + 0);
        float4 f1 = *(const float4*)(src + 4);
        unsigned short* dst = ldsA + ldrow * 32 + ldpart * 8;
        dst[0] = f2bf(f0.x); dst[1] = f2bf(f0.y);
        dst[2] = f2bf(f0.z); dst[3] = f2bf(f0.w);
        dst[4] = f2bf(f1.x); dst[5] = f2bf(f1.y);
        dst[6] = f2bf(f1.z); dst[7] = f2bf(f1.w);
        src += 32;
        if (kc + 1 < KCH) __builtin_prefetch(src, 0, 3);
        __syncthreads();

        v8u alo = *(const v8u*)(ldsA + arow * 32 + koff);
        v8u ahi = *(const v8u*)(ldsA + arow * 32 + 16 + koff);
        v16u au;
#pragma unroll
        for (int i = 0; i < 8; ++i) { au[i] = alo[i]; au[8 + i] = ahi[i]; }
        v16bf af = __builtin_bit_cast(v16bf, au);

#pragma unroll
        for (int j = 0; j < 8; ++j) {
            const int ntg = ntbase + j;
            v16u bu = *(const v16u*)(wmP + ((size_t)(ntg * 32 + kc) * 32 + lane) * 16);
            v16bf bf = __builtin_bit_cast(v16bf, bu);
            acc[j] = __builtin_amdgcn_wmma_f32_16x16x32_bf16(
                false, af, false, bf, (short)0, acc[j], false, false);
        }
        __syncthreads();
    }
#endif

    // Epilogue: C layout (lane = N col, VGPR r = M row, +8 for upper half-wave).
    float psum[8];
#pragma unroll
    for (int r = 0; r < 8; ++r) psum[r] = 0.0f;
#pragma unroll
    for (int j = 0; j < 8; ++j) {
        const int o  = o0 + (ngrp * 8 + j) * 16 + (lane & 15);
        const float gv = g[(size_t)b * HDIM + o];
        const float vv = v[o];
#pragma unroll
        for (int r = 0; r < 8; ++r)
            psum[r] += tanhf(acc[j][r] + gv) * vv;
    }
    const int trow = t0 + mt * 16 + ((lane < 16) ? 0 : 8);
#pragma unroll
    for (int r = 0; r < 8; ++r) {
        float s = psum[r];
        s += __shfl_xor(s, 1, 16);
        s += __shfl_xor(s, 2, 16);
        s += __shfl_xor(s, 4, 16);
        s += __shfl_xor(s, 8, 16);
        if ((lane & 15) == 0)
            atomicAdd(&sraw[(size_t)b * TLEN + trow + r], s);
    }
}

// ---------------------------------------------------------------------------
// Kernel 4: numerically-stable softmax over T per batch row.
// ---------------------------------------------------------------------------
__global__ __launch_bounds__(256)
void softmax_kernel(const float* __restrict__ sraw, float* __restrict__ score) {
    __shared__ float red[256];
    const int b   = blockIdx.x;
    const int tid = threadIdx.x;
    const float* s = sraw + (size_t)b * TLEN;

    float m = -INFINITY;
    for (int t = tid; t < TLEN; t += 256) m = fmaxf(m, s[t]);
    red[tid] = m; __syncthreads();
    for (int k = 128; k > 0; k >>= 1) {
        if (tid < k) red[tid] = fmaxf(red[tid], red[tid + k]);
        __syncthreads();
    }
    const float mx = red[0];
    __syncthreads();

    float sum = 0.0f;
    for (int t = tid; t < TLEN; t += 256) sum += expf(s[t] - mx);
    red[tid] = sum; __syncthreads();
    for (int k = 128; k > 0; k >>= 1) {
        if (tid < k) red[tid] += red[tid + k];
        __syncthreads();
    }
    const float inv = 1.0f / red[0];

    for (int t = tid; t < TLEN; t += 256)
        score[(size_t)b * TLEN + t] = expf(s[t] - mx) * inv;
}

// ---------------------------------------------------------------------------
// Kernel 5: c[b,h] = sum_t context[b,t,h] * score[b,t].
// Score row staged into LDS via TDM (1-D tile); context reads coalesced.
// ---------------------------------------------------------------------------
__global__ __launch_bounds__(256)
void wsum_kernel(const float* __restrict__ context,
                 const float* __restrict__ score,
                 float* __restrict__ c) {
    __shared__ float sl[TLEN];
    const int b = blockIdx.y;
    const int h = blockIdx.x * 256 + threadIdx.x;

#if HAVE_TDM
    if ((threadIdx.x >> 5) == 0) {
        tdm_load_2d((unsigned)(size_t)&sl[0], score + (size_t)b * TLEN,
                    /*tile*/TLEN, 0, /*tensor*/TLEN, 1u, /*stride0*/TLEN);
        __builtin_amdgcn_s_wait_tensorcnt(0);
    }
    __syncthreads();
#else
    for (int t = threadIdx.x; t < TLEN; t += 256)
        sl[t] = score[(size_t)b * TLEN + t];
    __syncthreads();
#endif

    const float* cb = context + (size_t)b * TLEN * HDIM + h;
    float a0 = 0.f, a1 = 0.f, a2 = 0.f, a3 = 0.f;
    for (int t = 0; t < TLEN; t += 4) {
        a0 += cb[(size_t)(t + 0) * HDIM] * sl[t + 0];
        a1 += cb[(size_t)(t + 1) * HDIM] * sl[t + 1];
        a2 += cb[(size_t)(t + 2) * HDIM] * sl[t + 2];
        a3 += cb[(size_t)(t + 3) * HDIM] * sl[t + 3];
        if (t + 8 < TLEN) __builtin_prefetch(cb + (size_t)(t + 8) * HDIM, 0, 3);
    }
    c[(size_t)b * HDIM + h] = (a0 + a1) + (a2 + a3);
}

// ---------------------------------------------------------------------------
// Host launcher.  Workspace layout (bytes):
//   [0, 128K)     g     : B*H fp32
//   [128K, 384K)  sraw  : B*T fp32
//   [384K, ~2.4M) wmP   : H*H bf16 (fragment-packed Wm)
// Output: c [B*H] then score [B*T], fp32.
// ---------------------------------------------------------------------------
extern "C" void kernel_launch(void* const* d_in, const int* in_sizes, int n_in,
                              void* d_out, int out_size, void* d_ws, size_t ws_size,
                              hipStream_t stream) {
    const float* x       = (const float*)d_in[0];
    const float* context = (const float*)d_in[1];
    // d_in[2] = mask (unused, matching reference)
    const float* Wq = (const float*)d_in[3];
    const float* bq = (const float*)d_in[4];
    const float* Wm = (const float*)d_in[5];
    const float* bm = (const float*)d_in[6];
    const float* v  = (const float*)d_in[7];

    float* out       = (float*)d_out;
    float* c_out     = out;                     // [B, H]
    float* score_out = out + BATCH * HDIM;      // [B, T]

    char* ws = (char*)d_ws;
    float*          g    = (float*)(ws + 0);
    float*          sraw = (float*)(ws + 131072);
    unsigned short* wmP  = (unsigned short*)(ws + 393216);

    zero_kernel<<<(BATCH * TLEN + 255) / 256, 256, 0, stream>>>(sraw, BATCH * TLEN);
    query_proj_kernel<<<dim3(HDIM / 8, BATCH), 256, 0, stream>>>(x, Wq, bq, bm, g);
    pack_wm_kernel<<<(HDIM * HDIM) / 256, 256, 0, stream>>>(Wm, wmP);
    gemm_tanh_kernel<<<dim3(TLEN / 64, HDIM / 256, BATCH), 256, 0, stream>>>(
        context, wmP, g, v, sraw);
    softmax_kernel<<<BATCH, 256, 0, stream>>>(sraw, score_out);
    wsum_kernel<<<dim3(HDIM / 256, BATCH), 256, 0, stream>>>(context, score_out, c_out);
}
